// MemoryEfficientISNEModel_45552423141377
// MI455X (gfx1250) — matrix operand order repaired and checked by reading, not compiled, not run on hardware
//
#include <hip/hip_runtime.h>
#include <hip/hip_bf16.h>
#include <math.h>

// ---------------------------------------------------------------------------
// Types for CDNA5 WMMA (wave32): A/B = 16 bf16 per lane, C/D = 8 f32 per lane.
// ---------------------------------------------------------------------------
typedef __attribute__((ext_vector_type(16))) __bf16 v16bf;
typedef __attribute__((ext_vector_type(8)))  float  v8f;

union BF16Frag { v16bf v; unsigned short u[16]; uint4 q[2]; };

__device__ __forceinline__ unsigned short f2bf(float f) {
    unsigned int u = __float_as_uint(f);
    u += 0x7FFFu + ((u >> 16) & 1u);     // round-to-nearest-even
    return (unsigned short)(u >> 16);
}
__device__ __forceinline__ float bf2f(unsigned short h) {
    return __uint_as_float(((unsigned int)h) << 16);
}

// ---------------------------------------------------------------------------
// h0 = node_features + emb_table[node_ids]   (N x 256, stored bf16)
// ---------------------------------------------------------------------------
__global__ void k_gather_embed(const int* __restrict__ ids,
                               const float* __restrict__ feat,
                               const float* __restrict__ emb,
                               unsigned short* __restrict__ out, int N) {
    int idx = blockIdx.x * blockDim.x + threadIdx.x;   // over N*256
    if (idx >= N * 256) return;
    int row = idx >> 8, c = idx & 255;
    float v = feat[idx] + emb[(long long)ids[row] * 256 + c];
    out[idx] = f2bf(v);
}

// ---------------------------------------------------------------------------
// Pack W (f32, KxNC row-major) into WMMA-B-fragment-major bf16:
// fragment f = (kk, ntile): 32 lanes x 16 contiguous bf16 (1 KB) so the GEMM
// reads each B fragment as two global_load_b128 (L2-resident).
// CDNA5 16-bit B layout: lanes 0-15 hold K kk*32+{0..7,16..23} (col = lane&15),
// lanes 16-31 hold K kk*32+{8..15,24..31}.
// ---------------------------------------------------------------------------
__global__ void k_pack_w(const float* __restrict__ W,
                         unsigned short* __restrict__ Wp, int K, int NC) {
    int o = blockIdx.x * blockDim.x + threadIdx.x;     // over K*NC
    if (o >= K * NC) return;
    int f    = o >> 9;            // 512 elements per fragment
    int rem  = o & 511;
    int lane = rem >> 4, j = rem & 15;
    int fragsPerK = NC >> 4;
    int kk = f / fragsPerK, nt = f - kk * fragsPerK;
    int k = kk * 32 + ((lane >> 4) << 3) + (j & 7) + ((j >> 3) << 4);
    int n = (nt << 4) + (lane & 15);
    Wp[o] = f2bf(W[(long long)k * NC + n]);
}

__global__ void k_zero_f32(float* __restrict__ p, int n) {
    int i = blockIdx.x * blockDim.x + threadIdx.x;
    if (i < n) p[i] = 0.0f;
}

// ---------------------------------------------------------------------------
// Fused GEMM (bf16 A[MxK] * packed-bf16 W) + bias + LayerNorm + optional ReLU.
// Block = 256 threads (8 waves), tile = 32 rows x NC cols.
// Each wave: 2 row-frags x TPW col-tiles -> 2*TPW WMMAs per k-step; no LDS /
// no barriers in the k-loop. B fragments read through a bumped pointer with
// constant immediate offsets. LayerNorm via width-16 shuffles + ds_add_f32.
// Epilogue stages the 32xNC tile in LDS and writes coalesced b128 stores.
// ---------------------------------------------------------------------------
template <int NC, bool OUTF32, bool RELU>
__global__ __launch_bounds__(256)
void k_gemm_bias_ln_act(const unsigned short* __restrict__ A,
                        const unsigned short* __restrict__ Wp,
                        const float* __restrict__ bias,
                        const float* __restrict__ gamma,
                        const float* __restrict__ beta,
                        void* __restrict__ outp,
                        int M, int K) {
    constexpr int TPW = NC / 128;        // col tiles per wave (8 waves): 4 or 2
    constexpr int FRAGS_PER_K = NC / 16; // B fragments per 32-deep k-slab
    constexpr int OSZ = OUTF32 ? 4 : 2;  // output element bytes

    __shared__ float s1S[32], s2S[32], muS[32], rsS[32];
    __shared__ __align__(16) unsigned char smO[32 * NC * OSZ];  // 32 KB tile

    const int tid  = threadIdx.x;
    const int lane = tid & 31;
    const int wave = tid >> 5;
    const int half = lane >> 4;          // 0: lanes 0-15, 1: lanes 16-31
    const int l16  = lane & 15;
    const int rowBase = blockIdx.x * 32;
    const int waveNT  = wave * TPW;      // first col-tile index for this wave

    if (tid < 32) { s1S[tid] = 0.0f; s2S[tid] = 0.0f; }
    __syncthreads();

    v8f acc[2][TPW];
    #pragma unroll
    for (int rf = 0; rf < 2; ++rf)
        #pragma unroll
        for (int t = 0; t < TPW; ++t)
            #pragma unroll
            for (int j = 0; j < 8; ++j) acc[rf][t][j] = 0.0f;

    // A fragment row pointers (tail-clamped). 16-bit A 16x32 layout:
    // lanes 0-15 -> K {0..7,16..23}, lanes 16-31 -> K {8..15,24..31}.
    const unsigned short* a0p =
        A + (long long)min(rowBase +      l16, M - 1) * K + half * 8;
    const unsigned short* a1p =
        A + (long long)min(rowBase + 16 + l16, M - 1) * K + half * 8;
    // B fragment pointer: lane-dependent base, bumped by a constant per step.
    const unsigned short* bp = Wp + ((size_t)waveNT * 32 + lane) * 16;
    constexpr size_t BSTEP = (size_t)FRAGS_PER_K * 512;  // elements per k-step

    const int kSteps = K >> 5;
    for (int kk = 0; kk < kSteps; ++kk) {
        BF16Frag a0, a1;
        a0.q[0] = ((const uint4*)a0p)[0];  // K +0..7
        a0.q[1] = ((const uint4*)a0p)[2];  // K +16..23
        a1.q[0] = ((const uint4*)a1p)[0];
        a1.q[1] = ((const uint4*)a1p)[2];
        a0p += 32; a1p += 32;

        #pragma unroll
        for (int t = 0; t < TPW; ++t) {
            BF16Frag bf_;
            const uint4* q = (const uint4*)(bp + t * 512);  // 1 KB apart
            bf_.q[0] = q[0];
            bf_.q[1] = q[1];
            acc[0][t] = __builtin_amdgcn_wmma_f32_16x16x32_bf16(
                false, a0.v, false, bf_.v, (short)0, acc[0][t], false, false);
            acc[1][t] = __builtin_amdgcn_wmma_f32_16x16x32_bf16(
                false, a1.v, false, bf_.v, (short)0, acc[1][t], false, false);
        }
        bp += BSTEP;
    }

    // Bias add (in-register) + per-row partial sums over this wave's cols.
    float bv[TPW];
    #pragma unroll
    for (int t = 0; t < TPW; ++t) bv[t] = bias[(waveNT + t) * 16 + l16];

    #pragma unroll
    for (int rf = 0; rf < 2; ++rf) {
        #pragma unroll
        for (int j = 0; j < 8; ++j) {
            float p1 = 0.0f, p2 = 0.0f;
            #pragma unroll
            for (int t = 0; t < TPW; ++t) {
                float y = acc[rf][t][j] + bv[t];
                acc[rf][t][j] = y;
                p1 += y; p2 += y * y;
            }
            #pragma unroll
            for (int o = 8; o > 0; o >>= 1) {   // reduce over l16 (same half)
                p1 += __shfl_xor(p1, o, 16);
                p2 += __shfl_xor(p2, o, 16);
            }
            if (l16 == 0) {
                const int row = rf * 16 + j + 8 * half;  // C/D row mapping
                atomicAdd(&s1S[row], p1);                // ds_add_f32
                atomicAdd(&s2S[row], p2);
            }
        }
    }
    __syncthreads();

    if (tid < 32) {
        float mu  = s1S[tid] / (float)NC;
        float var = s2S[tid] / (float)NC - mu * mu;
        muS[tid] = mu;
        rsS[tid] = rsqrtf(var + 1e-5f);
    }
    __syncthreads();

    // Normalize + affine + activation, stage tile in LDS.
    float gv[TPW], bev[TPW];
    #pragma unroll
    for (int t = 0; t < TPW; ++t) {
        const int col = (waveNT + t) * 16 + l16;
        gv[t]  = gamma[col];
        bev[t] = beta[col];
    }
    #pragma unroll
    for (int rf = 0; rf < 2; ++rf) {
        #pragma unroll
        for (int j = 0; j < 8; ++j) {
            const int row = rf * 16 + j + 8 * half;
            const float mu = muS[row], rs = rsS[row];
            #pragma unroll
            for (int t = 0; t < TPW; ++t) {
                const int col = (waveNT + t) * 16 + l16;
                float x = (acc[rf][t][j] - mu) * rs * gv[t] + bev[t];
                if (RELU) x = fmaxf(x, 0.0f);
                if (OUTF32) ((float*)smO)[row * NC + col] = x;
                else        ((unsigned short*)smO)[row * NC + col] = f2bf(x);
            }
        }
    }
    __syncthreads();

    // Coalesced b128 stores: the block's output is a contiguous 32xNC region.
    constexpr int VEC = 16 / OSZ;              // elements per uint4
    constexpr int TOTALV = 32 * NC / VEC;      // 2048 vectors either way
    const uint4* s4 = (const uint4*)smO;
    uint4* g4 = (uint4*)outp;
    const size_t gbase = ((size_t)rowBase * NC) / VEC;
    #pragma unroll
    for (int v = tid; v < TOTALV; v += 256) {
        const int row = (v * VEC) / NC;
        if (rowBase + row < M) g4[gbase + v] = s4[v];
    }
}

// ---------------------------------------------------------------------------
// Edge attention + scatter: one wave (32 lanes) per edge.
// att = sigmoid(h[row].wr + h[col].wc + b); agg[row] += att * h[col]
// ---------------------------------------------------------------------------
__global__ __launch_bounds__(256)
void k_edge_att_scatter(const int* __restrict__ ei,
                        const unsigned short* __restrict__ h,
                        const float* __restrict__ watt,
                        const float* __restrict__ batt,
                        float* __restrict__ agg, int E) {
    const int gw   = (blockIdx.x * 256 + threadIdx.x) >> 5;
    const int lane = threadIdx.x & 31;
    if (gw >= E) return;
    const int r = ei[gw], c = ei[E + gw];
    const unsigned short* hr = h + (long long)r * 512;
    const unsigned short* hc = h + (long long)c * 512;
    float s = 0.0f;
    #pragma unroll 4
    for (int j = lane; j < 512; j += 32)
        s += bf2f(hr[j]) * watt[j] + bf2f(hc[j]) * watt[512 + j];
    #pragma unroll
    for (int o = 16; o > 0; o >>= 1) s += __shfl_xor(s, o, 32);
    const float a = 1.0f / (1.0f + __expf(-(s + batt[0])));
    float* ag = agg + (long long)r * 512;
    for (int j = lane; j < 512; j += 32)
        atomicAdd(&ag[j], bf2f(hc[j]) * a);
}

// h_new = bf16( h2 + 0.5 * agg )
__global__ void k_combine(const unsigned short* __restrict__ h2,
                          const float* __restrict__ agg,
                          unsigned short* __restrict__ out, int n) {
    int i = blockIdx.x * blockDim.x + threadIdx.x;
    if (i < n) out[i] = f2bf(bf2f(h2[i]) + 0.5f * agg[i]);
}

// ---------------------------------------------------------------------------
// Launcher
// ---------------------------------------------------------------------------
extern "C" void kernel_launch(void* const* d_in, const int* in_sizes, int n_in,
                              void* d_out, int out_size, void* d_ws, size_t ws_size,
                              hipStream_t stream) {
    (void)n_in; (void)out_size; (void)ws_size;
    const int*   node_ids = (const int*)  d_in[0];
    const int*   edge_ix  = (const int*)  d_in[1];
    const float* feat     = (const float*)d_in[2];
    const float* emb      = (const float*)d_in[3];
    const float* W0  = (const float*)d_in[4];
    const float* b0  = (const float*)d_in[5];
    const float* g0  = (const float*)d_in[6];
    const float* be0 = (const float*)d_in[7];
    const float* W1  = (const float*)d_in[8];
    const float* b1  = (const float*)d_in[9];
    const float* g1  = (const float*)d_in[10];
    const float* be1 = (const float*)d_in[11];
    const float* W2  = (const float*)d_in[12];
    const float* b2  = (const float*)d_in[13];
    const float* g2  = (const float*)d_in[14];
    const float* be2 = (const float*)d_in[15];
    const float* W3  = (const float*)d_in[16];
    const float* b3  = (const float*)d_in[17];
    const float* g3  = (const float*)d_in[18];
    const float* be3 = (const float*)d_in[19];
    const float* watt = (const float*)d_in[20];
    const float* batt = (const float*)d_in[21];

    const int N = in_sizes[0];
    const int E = in_sizes[1] / 2;
    const int D = 256, H = 512;

    // Workspace carve-out (256-B aligned).
    char* ws = (char*)d_ws;
    auto carve = [&](size_t bytes) {
        char* p = ws;
        ws += (bytes + 255) & ~(size_t)255;
        return p;
    };
    unsigned short* W0p = (unsigned short*)carve((size_t)D * H * 2);
    unsigned short* W1p = (unsigned short*)carve((size_t)H * H * 2);
    unsigned short* W2p = (unsigned short*)carve((size_t)H * H * 2);
    unsigned short* W3p = (unsigned short*)carve((size_t)H * D * 2);
    unsigned short* h0b = (unsigned short*)carve((size_t)N * D * 2);
    unsigned short* hA  = (unsigned short*)carve((size_t)N * H * 2);
    unsigned short* hB  = (unsigned short*)carve((size_t)N * H * 2);
    float*          agg = (float*)         carve((size_t)N * H * 4);

    // One-time weight pack into WMMA-fragment-major bf16 (L2-resident after).
    k_pack_w<<<(D * H + 255) / 256, 256, 0, stream>>>(W0, W0p, D, H);
    k_pack_w<<<(H * H + 255) / 256, 256, 0, stream>>>(W1, W1p, H, H);
    k_pack_w<<<(H * H + 255) / 256, 256, 0, stream>>>(W2, W2p, H, H);
    k_pack_w<<<(H * D + 255) / 256, 256, 0, stream>>>(W3, W3p, H, D);

    // h0 = feat + emb[ids]
    k_gather_embed<<<(N * D + 255) / 256, 256, 0, stream>>>(node_ids, feat, emb, h0b, N);

    const int gemmBlocks = (N + 31) / 32;
    // Layer 0..2: GEMM + bias + LN + ReLU, bf16 out
    k_gemm_bias_ln_act<512, false, true><<<gemmBlocks, 256, 0, stream>>>(
        h0b, W0p, b0, g0, be0, hA, N, D);
    k_gemm_bias_ln_act<512, false, true><<<gemmBlocks, 256, 0, stream>>>(
        hA, W1p, b1, g1, be1, hB, N, H);
    k_gemm_bias_ln_act<512, false, true><<<gemmBlocks, 256, 0, stream>>>(
        hB, W2p, b2, g2, be2, hA, N, H);

    // Edge attention + scatter-sum
    k_zero_f32<<<(N * H + 255) / 256, 256, 0, stream>>>(agg, N * H);
    k_edge_att_scatter<<<((size_t)E * 32 + 255) / 256, 256, 0, stream>>>(
        edge_ix, hA, watt, batt, agg, E);

    // Residual, then final GEMM + bias + LN -> fp32 d_out (N x 256)
    k_combine<<<(N * H + 255) / 256, 256, 0, stream>>>(hA, agg, hB, N * H);
    k_gemm_bias_ln_act<256, true, false><<<gemmBlocks, 256, 0, stream>>>(
        hB, W3p, b3, g3, be3, d_out, N, H);
}
